// LikelihoodRatioEstimator_83674552861231
// MI455X (gfx1250) — matrix operand order repaired and checked by reading, not compiled, not run on hardware
//
#include <hip/hip_runtime.h>

#define NROWS 8192
#define DDIM  128

// LDS y-tile: 128 cols x 128 k, row stride padded 128->136 elements (TDM pad:
// 4 dwords after every 64 dwords) so ds_load_b128 lanes hit distinct banks.
#define LDS_STRIDE 136
#define LDS_HALF   (128 * LDS_STRIDE)            // elements per (hi|lo) plane
#define LDS_BYTES  (2 * LDS_HALF * 2)            // 69632 bytes

typedef __attribute__((ext_vector_type(16))) __bf16        v16bf;
typedef __attribute__((ext_vector_type(8)))  float         v8f;
typedef __attribute__((ext_vector_type(4)))  float         v4f;
typedef __attribute__((ext_vector_type(4)))  unsigned int  v4u;
typedef __attribute__((ext_vector_type(2)))  unsigned int  v2u;
typedef __attribute__((ext_vector_type(8)))  int           v8i;
typedef __attribute__((ext_vector_type(4)))  int           v4i;

union Frag {
    v16bf v;
    v4u   q[2];
};

__device__ __forceinline__ float waveRedSum(float v) {
    #pragma unroll
    for (int o = 16; o > 0; o >>= 1) v += __shfl_xor(v, o, 32);
    return v;
}

// Round-to-nearest-even bf16 split: x ~= hi + lo (~16 mantissa bits total)
__device__ __forceinline__ void split_bf16(float v, unsigned short& hi, unsigned short& lo) {
    unsigned int u  = __float_as_uint(v);
    unsigned int hb = (u + 0x7FFFu + ((u >> 16) & 1u)) & 0xFFFF0000u;
    hi = (unsigned short)(hb >> 16);
    float r = v - __uint_as_float(hb);
    unsigned int u2 = __float_as_uint(r);
    unsigned int lb = (u2 + 0x7FFFu + ((u2 >> 16) & 1u)) & 0xFFFF0000u;
    lo = (unsigned short)(lb >> 16);
}

// TDM: DMA a 128x128 (rows x k) 2-byte tile into LDS with bank-depad.
// D# per CDNA5 ISA ch.8: group0 {count, lds_addr, global_addr, type=2},
// group1 {data_size=2B, pad_enable, pad_interval=64dw, pad_amount=4dw,
//         tensor_dim0=128, tensor_dim1=128, tile_dim0=128, tile_dim1=128,
//         tensor_dim0_stride=128}, groups 2/3 zero (2D tile).
__device__ __forceinline__ void tdm_load_tile(const unsigned short* gsrc,
                                              unsigned int lds_byte_off) {
    unsigned long long ga = (unsigned long long)(uintptr_t)gsrc;
    v4u g0;
    g0[0] = 1u;                                   // count=1, user descriptor
    g0[1] = lds_byte_off;                         // lds_addr (bytes)
    g0[2] = (unsigned int)ga;                     // global_addr[31:0]
    g0[3] = ((unsigned int)(ga >> 32) & 0x01FFFFFFu) | 0x80000000u; // addr[56:32] | type=2
    v8i g1;
    g1[0] = 0x07510000;          // data_size=1(2B) | pad_en | pad_int=5(64dw) | pad_amt=3(4dw)
    g1[1] = (int)(128u << 16);   // tensor_dim0 = 128 (bits 79:48, low half)
    g1[2] = (int)(128u << 16);   // tensor_dim1 = 128 (bits 111:80, low half)
    g1[3] = (int)(128u << 16);   // tile_dim0   = 128 (bits 127:112)
    g1[4] = 128;                 // tile_dim1   = 128 (bits 143:128)
    g1[5] = 128;                 // tensor_dim0_stride = 128 (bits 191:160)
    g1[6] = 0;
    g1[7] = 0;
    v4i z4 = {0, 0, 0, 0};
    v8i z8 = {0, 0, 0, 0, 0, 0, 0, 0};
    __builtin_amdgcn_tensor_load_to_lds(g0, g1, z4, z4, z8, 0);
}

__global__ void zero_acc_kernel(double* acc) {
    if (threadIdx.x < 8) acc[threadIdx.x] = 0.0;
}

// Convert x,y to split-bf16 and compute fp32 row norms. One wave per row.
__global__ __launch_bounds__(256) void prep_kernel(
    const float* __restrict__ x, const float* __restrict__ y,
    unsigned short* __restrict__ xhi, unsigned short* __restrict__ xlo,
    unsigned short* __restrict__ yhi, unsigned short* __restrict__ ylo,
    float* __restrict__ x2, float* __restrict__ y2)
{
    const int gw   = blockIdx.x * 8 + (threadIdx.x >> 5);
    const int lane = threadIdx.x & 31;

    const float* src; unsigned short *dh, *dl; float* nrm; int row;
    if (gw < NROWS) { src = x; dh = xhi; dl = xlo; nrm = x2; row = gw; }
    else            { src = y; dh = yhi; dl = ylo; nrm = y2; row = gw - NROWS; }

    const v4f vv = *(const v4f*)(src + (size_t)row * DDIM + lane * 4);
    unsigned short h[4], l[4];
    float s = 0.0f;
    #pragma unroll
    for (int i = 0; i < 4; ++i) {
        s += vv[i] * vv[i];
        split_bf16(vv[i], h[i], l[i]);
    }
    v2u ph, pl;
    ph[0] = (unsigned int)h[0] | ((unsigned int)h[1] << 16);
    ph[1] = (unsigned int)h[2] | ((unsigned int)h[3] << 16);
    pl[0] = (unsigned int)l[0] | ((unsigned int)l[1] << 16);
    pl[1] = (unsigned int)l[2] | ((unsigned int)l[3] << 16);
    *(v2u*)(dh + (size_t)row * DDIM + lane * 4) = ph;
    *(v2u*)(dl + (size_t)row * DDIM + lane * 4) = pl;

    s = waveRedSum(s);
    if (lane == 0) nrm[row] = s;
}

// Tiled WMMA pass over the 8192x8192 logit matrix.
// mode 0: acc[0]+=sum q, acc[1]+=sum logit, acc[2]+=diag q, acc[3]+=diag logit
// mode 1: acc[4]+=sum q/(q+B), acc[5]+=diag q/(q+B)  (B from pass-0 results)
__global__ __launch_bounds__(256) void pass_kernel(
    const unsigned short* __restrict__ xhi, const unsigned short* __restrict__ xlo,
    const unsigned short* __restrict__ yhi, const unsigned short* __restrict__ ylo,
    const float* __restrict__ x2, const float* __restrict__ y2,
    double* acc, int mode)
{
    extern __shared__ unsigned short lds_buf[];   // dynamic LDS -> offset 0

    const int tid  = threadIdx.x;
    const int wv   = tid >> 5;
    const int lane = tid & 31;
    const int m16  = lane & 15;
    const int hf   = lane >> 4;

    const int rowBase = blockIdx.y * 128 + wv * 16;
    const int colBase = blockIdx.x * 128;
    const bool diagBlock = (blockIdx.x == blockIdx.y);

    // Stage this block's y-tile (hi+lo) into LDS via the Tensor Data Mover.
    if (wv == 0) {   // wave-uniform branch; TDM is a wave-level op
        tdm_load_tile(yhi + (size_t)colBase * DDIM, 0u);
        tdm_load_tile(ylo + (size_t)colBase * DDIM, (unsigned int)(LDS_HALF * 2));
        __builtin_amdgcn_s_wait_tensorcnt(0);
    }
    __syncthreads();

    float Bval = 0.0f;
    if (mode == 1) {
        const double M = (double)NROWS * (double)(NROWS - 1);
        Bval = (float)((acc[0] - acc[2]) / M);   // pass-0 results are final here
    }

    v8f c[8];
    #pragma unroll
    for (int t = 0; t < 8; ++t)
        #pragma unroll
        for (int r = 0; r < 8; ++r) c[t][r] = 0.0f;

    // A-fragment (16x32 bf16): lane holds row (m16), K pairs per ISA layout ->
    // two contiguous 16B global loads at k = kt*32 + hf*8 and +16.
    const int aRow = rowBase + m16;
    const unsigned short* aHiRow = xhi + (size_t)aRow * DDIM;
    const unsigned short* aLoRow = xlo + (size_t)aRow * DDIM;

    const unsigned short* lyhi = lds_buf;
    const unsigned short* lylo = lds_buf + LDS_HALF;

    #pragma unroll
    for (int kt = 0; kt < 4; ++kt) {
        const int ka = kt * 32 + hf * 8;
        Frag ahi, alo;
        ahi.q[0] = *(const v4u*)(aHiRow + ka);
        ahi.q[1] = *(const v4u*)(aHiRow + ka + 16);
        alo.q[0] = *(const v4u*)(aLoRow + ka);
        alo.q[1] = *(const v4u*)(aLoRow + ka + 16);

        const int kb = kt * 32 + hf * 8;
        #pragma unroll
        for (int ct = 0; ct < 8; ++ct) {
            // B-fragment (32x16): lane = column n, K at kb..kb+7 and kb+16..kb+23 from LDS.
            const int colLocal = ct * 16 + m16;
            const unsigned short* bH = lyhi + colLocal * LDS_STRIDE + kb;
            const unsigned short* bL = lylo + colLocal * LDS_STRIDE + kb;
            Frag bhi, blo;
            bhi.q[0] = *(const v4u*)(bH);
            bhi.q[1] = *(const v4u*)(bH + 16);
            blo.q[0] = *(const v4u*)(bL);
            blo.q[1] = *(const v4u*)(bL + 16);

            c[ct] = __builtin_amdgcn_wmma_f32_16x16x32_bf16(
                false, ahi.v, false, bhi.v, (short)0, c[ct], false, false);
            c[ct] = __builtin_amdgcn_wmma_f32_16x16x32_bf16(
                false, ahi.v, false, blo.v, (short)0, c[ct], false, false);
            c[ct] = __builtin_amdgcn_wmma_f32_16x16x32_bf16(
                false, alo.v, false, bhi.v, (short)0, c[ct], false, false);
        }
    }

    // Epilogue: d2 = max(|x_i|^2 + |y_j|^2 - 2 x.y, 0); q = 1/(1+d2) = exp(logit)
    float x2v[8];
    #pragma unroll
    for (int r = 0; r < 8; ++r) x2v[r] = x2[rowBase + r + hf * 8];

    float s0 = 0.0f, s1 = 0.0f, s2 = 0.0f, s3 = 0.0f;
    #pragma unroll
    for (int ct = 0; ct < 8; ++ct) {
        const int col = colBase + ct * 16 + m16;
        const float y2c = y2[col];
        #pragma unroll
        for (int r = 0; r < 8; ++r) {
            const int row = rowBase + r + hf * 8;
            const float dp = c[ct][r];
            const float d2 = fmaxf(x2v[r] + y2c - 2.0f * dp, 0.0f);
            const float q  = 1.0f / (1.0f + d2);
            if (mode == 0) {
                const float lg = -log1pf(d2);
                s0 += q; s1 += lg;
                if (row == col) { s2 += q; s3 += lg; }
            } else {
                const float sg = q / (q + Bval);
                s0 += sg;
                if (row == col) s2 += sg;
            }
        }
    }

    s0 = waveRedSum(s0);
    s2 = waveRedSum(s2);
    if (mode == 0) {
        s1 = waveRedSum(s1);
        s3 = waveRedSum(s3);
        if (lane == 0) {
            atomicAdd(&acc[0], (double)s0);
            atomicAdd(&acc[1], (double)s1);
            if (diagBlock) {
                atomicAdd(&acc[2], (double)s2);
                atomicAdd(&acc[3], (double)s3);
            }
        }
    } else {
        if (lane == 0) {
            atomicAdd(&acc[4], (double)s0);
            if (diagBlock) atomicAdd(&acc[5], (double)s2);
        }
    }
}

__global__ void finalize_kernel(const double* acc, float* out) {
    if (threadIdx.x != 0 || blockIdx.x != 0) return;
    const double Nd = (double)NROWS;
    const double M  = Nd * (Nd - 1.0);
    const double Soffq = acc[0] - acc[2];
    const double Soffl = acc[1] - acc[3];
    const double log_baseline = log(Soffq) - log(M);
    const double mean_pos = acc[3] / Nd - log_baseline;
    const double mean_neg = Soffl / M - log_baseline;
    const double attraction = -mean_pos;
    const double repulsion  = 1.0;  // mean(exp(neg - logmeanexp(neg))) == 1 analytically
    const double mean_sig_pos = acc[5] / Nd;
    const double mean_sig_neg = (acc[4] - acc[5]) / M;
    const double loss = attraction + repulsion;

    out[0] = (float)loss;
    out[1] = (float)mean_pos;
    out[2] = (float)mean_neg;
    out[3] = (float)mean_sig_pos;
    out[4] = (float)mean_sig_neg;
    out[5] = (float)attraction;
    out[6] = (float)repulsion;
    out[7] = (float)log_baseline;
}

extern "C" void kernel_launch(void* const* d_in, const int* in_sizes, int n_in,
                              void* d_out, int out_size, void* d_ws, size_t ws_size,
                              hipStream_t stream) {
    (void)in_sizes; (void)n_in; (void)out_size; (void)ws_size;
    const float* x = (const float*)d_in[0];   // context_embedding [N,D] fp32
    const float* y = (const float*)d_in[1];   // target_embedding  [N,D] fp32
    float* out = (float*)d_out;               // 8 fp32 scalars

    char* base = (char*)d_ws;
    double* acc = (double*)base;                                   // 8 doubles
    float*  x2  = (float*)(base + 64);                             // N floats
    float*  y2  = x2 + NROWS;                                      // N floats
    unsigned short* xhi = (unsigned short*)(base + 64 + (size_t)2 * NROWS * sizeof(float));
    unsigned short* xlo = xhi + (size_t)NROWS * DDIM;
    unsigned short* yhi = xlo + (size_t)NROWS * DDIM;
    unsigned short* ylo = yhi + (size_t)NROWS * DDIM;

    zero_acc_kernel<<<1, 32, 0, stream>>>(acc);
    prep_kernel<<<(2 * NROWS) / 8, 256, 0, stream>>>(x, y, xhi, xlo, yhi, ylo, x2, y2);

    dim3 grid(NROWS / 128, NROWS / 128);
    pass_kernel<<<grid, 256, LDS_BYTES, stream>>>(xhi, xlo, yhi, ylo, x2, y2, acc, 0);
    pass_kernel<<<grid, 256, LDS_BYTES, stream>>>(xhi, xlo, yhi, ylo, x2, y2, acc, 1);

    finalize_kernel<<<1, 32, 0, stream>>>(acc, out);
}
